// PointNetPlusCls_12824772346428
// MI455X (gfx1250) — matrix operand order repaired and checked by reading, not compiled, not run on hardware
//
#include <hip/hip_runtime.h>

typedef _Float16 f16;
typedef __attribute__((ext_vector_type(16))) _Float16 v16h;
typedef __attribute__((ext_vector_type(8)))  _Float16 v8h;
typedef __attribute__((ext_vector_type(8)))  float    v8f;

#define BN_EPS 1e-5f

// Build a v16h WMMA fragment from two contiguous 16-byte loads.
#define FRAG16(lo, hi) __builtin_shufflevector((lo), (hi), 0,1,2,3,4,5,6,7,8,9,10,11,12,13,14,15)

// ---------------- transpose (B,3,N) -> (B,N,3) ----------------
__global__ void k_transpose(const float* __restrict__ x, float* __restrict__ pts,
                            int B, int N) {
  int gid = blockIdx.x * blockDim.x + threadIdx.x;
  if (gid >= B * N) return;
  int b = gid / N, n = gid - b * N;
  const float* xb = x + (size_t)b * 3 * N;
  float* p = pts + (size_t)gid * 3;
  p[0] = xb[n];
  p[1] = xb[(size_t)N + n];
  p[2] = xb[(size_t)2 * N + n];
}

// ---------------- farthest point sampling: one block per batch ----------------
__global__ void k_fps(const float* __restrict__ pts, int N, int S, int* __restrict__ idx) {
  const int b = blockIdx.x;
  const int tid = threadIdx.x;
  const int NT = blockDim.x;
  __shared__ float dist[8192];      // 32 KB of the 320 KB WGP LDS pool
  __shared__ float red_v[256];
  __shared__ int   red_i[256];
  __shared__ int   cur;
  const float* P = pts + (size_t)b * N * 3;
  for (int i = tid; i < N; i += NT) dist[i] = 1.0e10f;
  if (tid == 0) cur = 0;
  __syncthreads();
  for (int s = 0; s < S; ++s) {
    int c = cur;
    if (tid == 0) idx[b * S + s] = c;
    float cx = P[c * 3 + 0], cy = P[c * 3 + 1], cz = P[c * 3 + 2];
    float bestv = -1.0f; int besti = 0;
    for (int i = tid; i < N; i += NT) {
      float dx = P[i * 3 + 0] - cx;
      float dy = P[i * 3 + 1] - cy;
      float dz = P[i * 3 + 2] - cz;
      float d = dx * dx + dy * dy + dz * dz;
      float nd = fminf(dist[i], d);
      dist[i] = nd;
      if (nd > bestv || (nd == bestv && i < besti)) { bestv = nd; besti = i; }
    }
    red_v[tid] = bestv; red_i[tid] = besti;
    __syncthreads();
    for (int off = NT >> 1; off > 0; off >>= 1) {
      if (tid < off) {
        float v2 = red_v[tid + off]; int i2 = red_i[tid + off];
        if (v2 > red_v[tid] || (v2 == red_v[tid] && i2 < red_i[tid])) {
          red_v[tid] = v2; red_i[tid] = i2;
        }
      }
      __syncthreads();
    }
    if (tid == 0) cur = red_i[0];
    __syncthreads();
  }
}

// ---------------- gather centroids: new_xyz[b,s] = pts[b, idx[b,s]] ----------------
__global__ void k_gather_xyz(const float* __restrict__ src, const int* __restrict__ idx,
                             float* __restrict__ dst, int B, int N, int S) {
  int gid = blockIdx.x * blockDim.x + threadIdx.x;
  if (gid >= B * S) return;
  int b = gid / S;
  int p = idx[gid];
  const float* sp = src + ((size_t)b * N + p) * 3;
  float* dp = dst + (size_t)gid * 3;
  dp[0] = sp[0]; dp[1] = sp[1]; dp[2] = sp[2];
}

// ---------------- ball query: one wave32 per centroid (ballot compaction) ------
__global__ void k_ballquery(const float* __restrict__ sxyz, const float* __restrict__ nxyz,
                            int B, int N, int S, int K, float r2,
                            int* __restrict__ gidx) {
  int gt = blockIdx.x * blockDim.x + threadIdx.x;
  int wid = gt >> 5;
  int lane = gt & 31;
  if (wid >= B * S) return;
  int b = wid / S;
  const float* q = nxyz + (size_t)wid * 3;
  float qx = q[0], qy = q[1], qz = q[2];
  const float* P = sxyz + (size_t)b * N * 3;
  int* out = gidx + (size_t)wid * K;
  int cnt = 0;
  int first = -1;
  for (int base = 0; base < N; base += 32) {
    if (cnt >= K) break;
    int i = base + lane;
    bool inball = false;
    if (i < N) {
      float dx = P[i * 3 + 0] - qx;
      float dy = P[i * 3 + 1] - qy;
      float dz = P[i * 3 + 2] - qz;
      inball = (dx * dx + dy * dy + dz * dz) <= r2;
    }
    unsigned mask = __builtin_amdgcn_ballot_w32(inball);
    if (first < 0 && mask != 0u) first = base + __builtin_ctz(mask);
    int pos = cnt + __builtin_popcount(mask & ((1u << lane) - 1u));
    if (inball && pos < K) out[pos] = i;
    cnt += __builtin_popcount(mask);
  }
  if (cnt > K) cnt = K;
  for (int p = cnt + lane; p < K; p += 32) out[p] = first;
}

// ---------------- gather+center+concat into padded f16 activation matrix ------
__global__ void k_group(const float* __restrict__ sxyz, const f16* __restrict__ feats,
                        const float* __restrict__ nxyz, const int* __restrict__ gidx,
                        f16* __restrict__ out,
                        int B, int N, int S, int Kg, int Cf, int Cpad) {
  size_t gid = (size_t)blockIdx.x * blockDim.x + threadIdx.x;
  size_t total = (size_t)B * S * Kg * Cpad;
  if (gid >= total) return;
  int c = (int)(gid % Cpad);
  size_t row = gid / Cpad;
  int k = (int)(row % Kg);
  size_t bs = row / Kg;            // b*S + s
  int b = (int)(bs / S);
  int pt = gidx[bs * Kg + k];
  float v = 0.0f;
  if (c < 3) {
    v = sxyz[((size_t)b * N + pt) * 3 + c] - nxyz[bs * 3 + c];
  } else if (c < 3 + Cf) {
    v = (float)feats[((size_t)b * N + pt) * Cf + (c - 3)];
  }
  out[gid] = (f16)v;
}

// ---------------- weight pre-pack: f32 (COsrc x CIN) -> f16 (COdst x KP), zero pad
__global__ void k_packw(const float* __restrict__ W, f16* __restrict__ Wh,
                        int COdst, int COsrc, int CIN, int KP) {
  int gid = blockIdx.x * blockDim.x + threadIdx.x;
  if (gid >= COdst * KP) return;
  int k = gid % KP;
  int o = gid / KP;
  float v = (o < COsrc && k < CIN) ? W[(size_t)o * CIN + k] : 0.0f;
  Wh[gid] = (f16)v;
}

// ---------------- WMMA GEMM + folded BN + ReLU -------------------------------
// O[m, n] = relu( (sum_c A[m,c]*Wh[n,c] + bias[n] - rm[n]) * gamma/sqrt(rv+eps) + beta[n] )
// A: M x KP row-major f16. Wh: CO x KP row-major f16 (pre-padded).
// One wave computes a 16(M) x 64(N) strip: A fragment reused across 4 WMMAs.
__global__ void k_wmma_mlp(const f16* __restrict__ A, const f16* __restrict__ Wh,
                           const float* __restrict__ bias, const float* __restrict__ gamma,
                           const float* __restrict__ beta, const float* __restrict__ rm,
                           const float* __restrict__ rvv,
                           f16* __restrict__ O, int M, int KP, int CO) {
  const int lane = threadIdx.x & 31;
  const int wave = threadIdx.x >> 5;
  const int mt = blockIdx.x;
  const int ng = blockIdx.y * (blockDim.x >> 5) + wave;  // 64-wide N group
  if (ng * 64 >= CO) return;            // uniform per wave: EXEC all-1s for WMMA
  const int row  = lane & 15;
  const int half = lane >> 4;
  const int m0 = mt * 16;
  const int n0 = ng * 64 + row;         // B-matrix column / output channel, tile 0
  // A fragment per lane: K = [half*8 .. +7] and [16+half*8 .. +7]  (two b128 loads)
  const f16* arow = A + (size_t)(m0 + row) * KP + (half << 3);
  // B fragment per lane: K = [half*8 .. +7] and [16+half*8 .. +7]  (two b128 loads)
  const f16* w0 = Wh + (size_t)(n0     ) * KP + (half << 3);
  const f16* w1 = Wh + (size_t)(n0 + 16) * KP + (half << 3);
  const f16* w2 = Wh + (size_t)(n0 + 32) * KP + (half << 3);
  const f16* w3 = Wh + (size_t)(n0 + 48) * KP + (half << 3);
  v8f acc0 = {}, acc1 = {}, acc2 = {}, acc3 = {};
  for (int k0 = 0; k0 < KP; k0 += 32) {
    v16h a = FRAG16(*(const v8h*)(arow + k0), *(const v8h*)(arow + k0 + 16));
    __builtin_prefetch(arow + k0 + 128, 0, 1);   // global_prefetch_b8
    v16h b0 = FRAG16(*(const v8h*)(w0 + k0), *(const v8h*)(w0 + k0 + 16));
    v16h b1 = FRAG16(*(const v8h*)(w1 + k0), *(const v8h*)(w1 + k0 + 16));
    v16h b2 = FRAG16(*(const v8h*)(w2 + k0), *(const v8h*)(w2 + k0 + 16));
    v16h b3 = FRAG16(*(const v8h*)(w3 + k0), *(const v8h*)(w3 + k0 + 16));
    acc0 = __builtin_amdgcn_wmma_f32_16x16x32_f16(false, a, false, b0, (short)0, acc0, false, false);
    acc1 = __builtin_amdgcn_wmma_f32_16x16x32_f16(false, a, false, b1, (short)0, acc1, false, false);
    acc2 = __builtin_amdgcn_wmma_f32_16x16x32_f16(false, a, false, b2, (short)0, acc2, false, false);
    acc3 = __builtin_amdgcn_wmma_f32_16x16x32_f16(false, a, false, b3, (short)0, acc3, false, false);
  }
#pragma unroll
  for (int t = 0; t < 4; ++t) {
    const v8f acc = (t == 0) ? acc0 : (t == 1) ? acc1 : (t == 2) ? acc2 : acc3;
    int o = n0 + t * 16;
    float sc = gamma[o] * rsqrtf(rvv[o] + BN_EPS);
    float sh = (bias[o] - rm[o]) * sc + beta[o];
#pragma unroll
    for (int i = 0; i < 8; ++i) {
      float y = fmaxf(acc[i] * sc + sh, 0.0f);
      int m = m0 + i + (half << 3);     // D layout: VGPR i -> M = i + 8*half
      O[(size_t)m * CO + o] = (f16)y;
    }
  }
}

// ---------------- WMMA classifier head: 16x40 = (16x256) @ (256x40)^T --------
// Wh pre-packed to 48 x 256 f16 (rows 40..47 zero). One block, 3 waves.
__global__ void k_wmma_head(const f16* __restrict__ A, const f16* __restrict__ Wh,
                            const float* __restrict__ bias, float* __restrict__ out) {
  const int lane = threadIdx.x & 31;
  const int nt = threadIdx.x >> 5;      // 0..2
  const int row = lane & 15;
  const int half = lane >> 4;
  const int KP = 256;
  const f16* arow = A + (size_t)row * KP + (half << 3);
  const f16* wr = Wh + (size_t)(nt * 16 + row) * KP + (half << 3);
  v8f acc = {};
  for (int k0 = 0; k0 < KP; k0 += 32) {
    v16h a = FRAG16(*(const v8h*)(arow + k0), *(const v8h*)(arow + k0 + 16));
    v16h b = FRAG16(*(const v8h*)(wr + k0), *(const v8h*)(wr + k0 + 16));
    acc = __builtin_amdgcn_wmma_f32_16x16x32_f16(false, a, false, b, (short)0, acc, false, false);
  }
  int o = nt * 16 + row;
  if (o < 40) {
    float bb = bias[o];
#pragma unroll
    for (int i = 0; i < 8; ++i) {
      int m = i + (half << 3);
      out[(size_t)m * 40 + o] = acc[i] + bb;
    }
  }
}

// ---------------- max-pool over the group dimension --------------------------
__global__ void k_maxpool(const f16* __restrict__ H, f16* __restrict__ O,
                          int BS, int Kg, int C) {
  int gid = blockIdx.x * blockDim.x + threadIdx.x;
  if (gid >= BS * C) return;
  int ch = gid % C;
  int bs = gid / C;
  const f16* h = H + (size_t)bs * Kg * C + ch;
  float m = -3.0e38f;
  for (int k = 0; k < Kg; ++k) m = fmaxf(m, (float)h[(size_t)k * C]);
  O[gid] = (f16)m;
}

extern "C" void kernel_launch(void* const* d_in, const int* in_sizes, int n_in,
                              void* d_out, int out_size, void* d_ws, size_t ws_size,
                              hipStream_t stream) {
  (void)in_sizes; (void)n_in; (void)out_size; (void)ws_size;
  const int B = 16, N0 = 8192;

  // d_in layout (setup_inputs dict order; each param dict = W,b,gamma,beta,rm,rv):
  // [0] x | [1..18] sa1 (3 layers) | [19..36] sa2 | [37..54] sa3 | [55..66] fc (2 layers)
  // [67] head_w | [68] head_b
  const int SA1 = 1, SA2 = 19, SA3 = 37, FC = 55;
  const float* x      = (const float*)d_in[0];
  const float* head_w = (const float*)d_in[67];
  const float* head_b = (const float*)d_in[68];

  char* ws = (char*)d_ws;
  size_t cur = 0;
  auto take = [&](size_t bytes) -> void* {
    void* p = (void*)(ws + cur);
    cur += (bytes + 255) & ~(size_t)255;
    return p;
  };
  float* pts  = (float*)take((size_t)B * N0 * 3 * 4);
  float* xyz1 = (float*)take((size_t)B * 512 * 3 * 4);
  float* xyz2 = (float*)take((size_t)B * 128 * 3 * 4);
  float* xyz3 = (float*)take((size_t)B * 1 * 3 * 4);
  int*   sidx = (int*)take((size_t)B * 512 * 4);
  int*   gidx = (int*)take((size_t)B * 512 * 32 * 4);
  f16*   f1   = (f16*)take((size_t)B * 512 * 128 * 2);
  f16*   f2   = (f16*)take((size_t)B * 128 * 256 * 2);
  f16*   f3   = (f16*)take((size_t)B * 1024 * 2);
  f16*   fcA  = (f16*)take((size_t)B * 512 * 2);
  f16*   fcB  = (f16*)take((size_t)B * 256 * 2);
  f16*   wbuf = (f16*)take((size_t)1024 * 1024 * 2);   // packed f16 weights (reused)
  f16*   bufA = (f16*)take((size_t)70 * 1024 * 1024);  // ping
  f16*   bufB = (f16*)take((size_t)70 * 1024 * 1024);  // pong

  auto cdiv = [](long long a, long long b) -> int { return (int)((a + b - 1) / b); };

  auto gemm = [&](const f16* Ain, int pbase, f16* Out, int M, int CIN, int KP, int CO) {
    const float* Wp  = (const float*)d_in[pbase + 0];
    const float* bi  = (const float*)d_in[pbase + 1];
    const float* ga  = (const float*)d_in[pbase + 2];
    const float* be  = (const float*)d_in[pbase + 3];
    const float* rmp = (const float*)d_in[pbase + 4];
    const float* rvp = (const float*)d_in[pbase + 5];
    k_packw<<<cdiv((long long)CO * KP, 256), 256, 0, stream>>>(Wp, wbuf, CO, CO, CIN, KP);
    dim3 grid(M / 16, cdiv(cdiv(CO, 64), 4));  // 4 waves per block, one 16x64 strip each
    k_wmma_mlp<<<grid, 128, 0, stream>>>(Ain, wbuf, bi, ga, be, rmp, rvp, Out, M, KP, CO);
  };

  // ---- input transpose ----
  k_transpose<<<cdiv((long long)B * N0, 256), 256, 0, stream>>>(x, pts, B, N0);

  // ---- SA1: N=8192 -> S=512, K=32, r=0.2, MLP 3->64->64->128 ----
  {
    const int N = N0, S = 512, Kg = 32, Cpad = 32;
    const int M = B * S * Kg;  // 262144
    k_fps<<<B, 256, 0, stream>>>(pts, N, S, sidx);
    k_gather_xyz<<<cdiv((long long)B * S, 256), 256, 0, stream>>>(pts, sidx, xyz1, B, N, S);
    k_ballquery<<<cdiv((long long)B * S * 32, 256), 256, 0, stream>>>(pts, xyz1, B, N, S, Kg, 0.04f, gidx);
    k_group<<<cdiv((long long)M * Cpad, 256), 256, 0, stream>>>(pts, (const f16*)nullptr, xyz1, gidx, bufA, B, N, S, Kg, 0, Cpad);
    gemm(bufA, SA1 + 0,  bufB, M, 3,   32,  64);
    gemm(bufB, SA1 + 6,  bufA, M, 64,  64,  64);
    gemm(bufA, SA1 + 12, bufB, M, 64,  64,  128);
    k_maxpool<<<cdiv((long long)B * S * 128, 256), 256, 0, stream>>>(bufB, f1, B * S, Kg, 128);
  }

  // ---- SA2: N=512 -> S=128, K=64, r=0.4, MLP 131->128->128->256 ----
  {
    const int N = 512, S = 128, Kg = 64, Cf = 128, Cpad = 160;
    const int M = B * S * Kg;  // 131072
    k_fps<<<B, 256, 0, stream>>>(xyz1, N, S, sidx);
    k_gather_xyz<<<cdiv((long long)B * S, 256), 256, 0, stream>>>(xyz1, sidx, xyz2, B, N, S);
    k_ballquery<<<cdiv((long long)B * S * 32, 256), 256, 0, stream>>>(xyz1, xyz2, B, N, S, Kg, 0.16f, gidx);
    k_group<<<cdiv((long long)M * Cpad, 256), 256, 0, stream>>>(xyz1, f1, xyz2, gidx, bufA, B, N, S, Kg, Cf, Cpad);
    gemm(bufA, SA2 + 0,  bufB, M, 131, 160, 128);
    gemm(bufB, SA2 + 6,  bufA, M, 128, 128, 128);
    gemm(bufA, SA2 + 12, bufB, M, 128, 128, 256);
    k_maxpool<<<cdiv((long long)B * S * 256, 256), 256, 0, stream>>>(bufB, f2, B * S, Kg, 256);
  }

  // ---- SA3: N=128 -> S=1, K=128, r=0.8, MLP 259->256->512->1024 ----
  {
    const int N = 128, S = 1, Kg = 128, Cf = 256, Cpad = 288;
    const int M = B * S * Kg;  // 2048
    k_fps<<<B, 256, 0, stream>>>(xyz2, N, S, sidx);
    k_gather_xyz<<<1, 256, 0, stream>>>(xyz2, sidx, xyz3, B, N, S);
    k_ballquery<<<cdiv((long long)B * S * 32, 256), 256, 0, stream>>>(xyz2, xyz3, B, N, S, Kg, 0.64f, gidx);
    k_group<<<cdiv((long long)M * Cpad, 256), 256, 0, stream>>>(xyz2, f2, xyz3, gidx, bufA, B, N, S, Kg, Cf, Cpad);
    gemm(bufA, SA3 + 0,  bufB, M, 259, 288, 256);
    gemm(bufB, SA3 + 6,  bufA, M, 256, 256, 512);
    gemm(bufA, SA3 + 12, bufB, M, 512, 512, 1024);
    k_maxpool<<<cdiv((long long)B * 1024, 256), 256, 0, stream>>>(bufB, f3, B, Kg, 1024);
  }

  // ---- FC 1024->512->256 (M = B = 16) + WMMA head 256->40 ----
  gemm(f3,  FC + 0, fcA, 16, 1024, 1024, 512);
  gemm(fcA, FC + 6, fcB, 16, 512,  512,  256);
  k_packw<<<cdiv((long long)48 * 256, 256), 256, 0, stream>>>(head_w, wbuf, 48, 40, 256, 256);
  k_wmma_head<<<1, 96, 0, stream>>>(fcB, wbuf, head_b, (float*)d_out);
}
